// Encoding_27444841021727
// MI455X (gfx1250) — matrix operand order repaired and checked by reading, not compiled, not run on hardware
//
#include <hip/hip_runtime.h>

#define BB 64
#define LL 1024
#define DD 448
#define D2 896

typedef __attribute__((ext_vector_type(16))) __bf16 v16bf;
typedef __attribute__((ext_vector_type(8)))  float  v8f;

union Frag { v16bf v; uint4 q[2]; };

__device__ __forceinline__ unsigned short f2bf(float f) {
    unsigned int u = __float_as_uint(f);
    unsigned int r = u + 0x7FFFu + ((u >> 16) & 1u);   // round-to-nearest-even
    return (unsigned short)(r >> 16);
}

// A-fragment (16x32 bf16, M=row per lane%16): lane-half selects K pattern
// elems 0..7  -> K = k0 + 8*half + e ; elems 8..15 -> K = k0 + 16 + 8*half + e
__device__ __forceinline__ v16bf load_a(const unsigned short* __restrict__ row, int k0, int half) {
    Frag f;
    f.q[0] = *(const uint4*)(row + k0 + 8 * half);
    f.q[1] = *(const uint4*)(row + k0 + 16 + 8 * half);
    return f.v;
}

// B-fragment (32x16 bf16, N=col per lane%16): elems 0..7 -> K = k0 + 8*half + e ; elems 8..15 -> K = k0 + 16 + 8*half + e
__device__ __forceinline__ v16bf load_b(const unsigned short* __restrict__ row, int k0, int half) {
    Frag f;
    f.q[0] = *(const uint4*)(row + k0 + 8 * half);
    f.q[1] = *(const uint4*)(row + k0 + 16 + 8 * half);
    return f.v;
}

#define WMMA_BF16(a, b, c) \
    __builtin_amdgcn_wmma_f32_16x16x32_bf16(false, (a), false, (b), (short)0, (c), false, false)

// ---------------- prep: bf16 convert, x*wc, sa/sb row dots -------------------
__global__ void __launch_bounds__(256) prep_rows_kernel(
    const float* __restrict__ x, const float* __restrict__ w_itr,
    unsigned short* __restrict__ x_bf, unsigned short* __restrict__ xwc_bf,
    float* __restrict__ sa, float* __restrict__ sb)
{
    const int wave = threadIdx.x >> 5, lane = threadIdx.x & 31;
    const size_t row = (size_t)blockIdx.x * 8 + wave;          // < B*L
    const float* xr = x + row * DD;
    float asum = 0.f, bsum = 0.f;
    #pragma unroll
    for (int t = 0; t < DD / 32; ++t) {
        int d = t * 32 + lane;
        float v = xr[d];
        x_bf[row * DD + d]   = f2bf(v);
        xwc_bf[row * DD + d] = f2bf(v * w_itr[2 * DD + d]);
        asum += v * w_itr[d];
        bsum += v * w_itr[DD + d];
    }
    #pragma unroll
    for (int o = 16; o >= 1; o >>= 1) {
        asum += __shfl_xor(asum, o, 32);
        bsum += __shfl_xor(bsum, o, 32);
    }
    if (lane == 0) { sa[row] = asum; sb[row] = bsum; }
}

// ---------------- x transpose -> xT_bf [B, D, L] -----------------------------
__global__ void __launch_bounds__(256) transpose_x_kernel(
    const float* __restrict__ x, unsigned short* __restrict__ xT_bf)
{
    __shared__ float tile[32][33];
    const int b = blockIdx.z;
    const int d0 = blockIdx.x * 32, l0 = blockIdx.y * 32;
    const int tx = threadIdx.x, ty = threadIdx.y;              // 32 x 8
    #pragma unroll
    for (int i = 0; i < 32; i += 8)
        tile[ty + i][tx] = x[((size_t)b * LL + l0 + ty + i) * DD + d0 + tx];
    __syncthreads();
    #pragma unroll
    for (int i = 0; i < 32; i += 8)
        xT_bf[((size_t)b * DD + d0 + ty + i) * LL + l0 + tx] = f2bf(tile[tx][ty + i]);
}

// ---------------- weight transpose -> wT [D, 2D] bf16 ------------------------
__global__ void __launch_bounds__(256) prep_w_kernel(
    const float* __restrict__ w, unsigned short* __restrict__ wT)
{
    int idx = blockIdx.x * 256 + threadIdx.x;                  // over 2D*D
    if (idx < D2 * DD) {
        int k = idx / DD, n = idx % DD;
        wT[(size_t)n * D2 + k] = f2bf(w[idx]);
    }
}

// ---------------- S^T + softmax -> attn_bf (unnormalized), inv_l -------------
__global__ void __launch_bounds__(256) attn_softmax_kernel(
    const unsigned short* __restrict__ x_bf, const unsigned short* __restrict__ xwc_bf,
    const float* __restrict__ sa, const float* __restrict__ sb,
    unsigned short* __restrict__ attn_bf, float* __restrict__ inv_l)
{
    const int wave = threadIdx.x >> 5, lane = threadIdx.x & 31;
    const int half = lane >> 4, ln = lane & 15;
    const int t = blockIdx.x * 8 + wave;                       // < B*(L/16)
    const int b = t >> 6;
    const int i0 = (t & 63) << 4;
    const int i = i0 + ln;
    const float sai = sa[b * LL + i];
    const float* sbp = sb + b * LL;
    const unsigned short* xb  = x_bf + (size_t)b * LL * DD;
    const unsigned short* xwb = xwc_bf + ((size_t)b * LL + i) * DD;

    float m = -1e30f;
    for (int j0 = 0; j0 < LL; j0 += 16) {
        v8f c = {};
        const unsigned short* arow = xb + (size_t)(j0 + ln) * DD;
        #pragma unroll
        for (int k0 = 0; k0 < DD; k0 += 32) {
            v16bf a  = load_a(arow, k0, half);
            v16bf bm = load_b(xwb, k0, half);
            c = WMMA_BF16(a, bm, c);
        }
        #pragma unroll
        for (int r = 0; r < 8; ++r)
            m = fmaxf(m, c[r] + sai + sbp[j0 + 8 * half + r]);
    }
    m = fmaxf(m, __shfl_xor(m, 16, 32));

    float lsum = 0.f;
    unsigned short* orow = attn_bf + ((size_t)b * LL + i) * LL;
    for (int j0 = 0; j0 < LL; j0 += 16) {
        v8f c = {};
        const unsigned short* arow = xb + (size_t)(j0 + ln) * DD;
        #pragma unroll
        for (int k0 = 0; k0 < DD; k0 += 32) {
            v16bf a  = load_a(arow, k0, half);
            v16bf bm = load_b(xwb, k0, half);
            c = WMMA_BF16(a, bm, c);
        }
        union { uint4 q; unsigned short u[8]; } st;
        #pragma unroll
        for (int r = 0; r < 8; ++r) {
            float p = __expf(c[r] + sai + sbp[j0 + 8 * half + r] - m);
            lsum += p;
            st.u[r] = f2bf(p);
        }
        *(uint4*)(orow + j0 + 8 * half) = st.q;                // 8 consecutive j
    }
    lsum += __shfl_xor(lsum, 16, 32);
    if (half == 0) inv_l[b * LL + i] = 1.0f / lsum;
}

// ------- itr^T = x^T @ attn^T, register-blocked 4 d-tiles x 2 i-tiles --------
__global__ void __launch_bounds__(256) attn_apply_kernel(
    const unsigned short* __restrict__ xT_bf, const unsigned short* __restrict__ attn_bf,
    const float* __restrict__ inv_l, unsigned short* __restrict__ itr_bf)
{
    const int wave = threadIdx.x >> 5, lane = threadIdx.x & 31;
    const int half = lane >> 4, ln = lane & 15;
    const int t = blockIdx.x * 8 + wave;                       // < B * 7 * 32
    const int b   = t / 224;
    const int rem = t % 224;
    const int d0  = (rem / 32) * 64;                           // 4 d-tiles
    const int i0  = (rem % 32) * 32;                           // 2 i-tiles

    const unsigned short* arow[4];
    const unsigned short* brow[2];
    #pragma unroll
    for (int dd = 0; dd < 4; ++dd)
        arow[dd] = xT_bf + ((size_t)b * DD + d0 + 16 * dd + ln) * LL;
    #pragma unroll
    for (int ii = 0; ii < 2; ++ii)
        brow[ii] = attn_bf + ((size_t)b * LL + i0 + 16 * ii + ln) * LL;

    v8f c[4][2] = {};
    for (int j0 = 0; j0 < LL; j0 += 32) {
        v16bf bm[2];
        #pragma unroll
        for (int ii = 0; ii < 2; ++ii) bm[ii] = load_b(brow[ii], j0, half);
        #pragma unroll
        for (int dd = 0; dd < 4; ++dd) {
            v16bf a = load_a(arow[dd], j0, half);
            #pragma unroll
            for (int ii = 0; ii < 2; ++ii)
                c[dd][ii] = WMMA_BF16(a, bm[ii], c[dd][ii]);
        }
    }
    #pragma unroll
    for (int ii = 0; ii < 2; ++ii) {
        const int i = i0 + 16 * ii + ln;
        const float scale = inv_l[b * LL + i];
        unsigned short* orow = itr_bf + ((size_t)b * LL + i) * DD;
        #pragma unroll
        for (int dd = 0; dd < 4; ++dd) {
            union { uint4 q; unsigned short u[8]; } st;
            #pragma unroll
            for (int r = 0; r < 8; ++r) st.u[r] = f2bf(c[dd][ii][r] * scale);
            *(uint4*)(orow + d0 + 16 * dd + 8 * half) = st.q;
        }
    }
}

// -------- fused MLP + gates, register-blocked 2 l-tiles x 3 gates ------------
__global__ void __launch_bounds__(256) mlp_kernel(
    const float* __restrict__ x,
    const unsigned short* __restrict__ x_bf, const unsigned short* __restrict__ itr_bf,
    const unsigned short* __restrict__ w1T, const unsigned short* __restrict__ w2T,
    const unsigned short* __restrict__ w3T,
    const float* __restrict__ b1, const float* __restrict__ b2, const float* __restrict__ b3,
    float* __restrict__ out)
{
    const int wave = threadIdx.x >> 5, lane = threadIdx.x & 31;
    const int half = lane >> 4, ln = lane & 15;
    const int t = blockIdx.x * 8 + wave;                       // < B * 32 * 28
    const int b   = t / 896;
    const int rem = t % 896;
    const int l0  = (rem / 28) * 32;                           // 2 l-tiles
    const int d0  = (rem % 28) * 16;

    const unsigned short* xrow[2];
    const unsigned short* irow[2];
    #pragma unroll
    for (int lt = 0; lt < 2; ++lt) {
        xrow[lt] = x_bf   + ((size_t)b * LL + l0 + 16 * lt + ln) * DD;
        irow[lt] = itr_bf + ((size_t)b * LL + l0 + 16 * lt + ln) * DD;
    }
    const unsigned short* w1r = w1T + (size_t)(d0 + ln) * D2;
    const unsigned short* w2r = w2T + (size_t)(d0 + ln) * D2;
    const unsigned short* w3r = w3T + (size_t)(d0 + ln) * D2;

    v8f c1[2] = {}, c2[2] = {}, c3[2] = {};
    #pragma unroll 2
    for (int k0 = 0; k0 < DD; k0 += 32) {                      // h[:, :D] = x
        v16bf bb1 = load_b(w1r, k0, half);
        v16bf bb2 = load_b(w2r, k0, half);
        v16bf bb3 = load_b(w3r, k0, half);
        #pragma unroll
        for (int lt = 0; lt < 2; ++lt) {
            v16bf a = load_a(xrow[lt], k0, half);
            c1[lt] = WMMA_BF16(a, bb1, c1[lt]);
            c2[lt] = WMMA_BF16(a, bb2, c2[lt]);
            c3[lt] = WMMA_BF16(a, bb3, c3[lt]);
        }
    }
    #pragma unroll 2
    for (int k0 = 0; k0 < DD; k0 += 32) {                      // h[:, D:] = itr
        v16bf bb1 = load_b(w1r, k0 + DD, half);
        v16bf bb2 = load_b(w2r, k0 + DD, half);
        v16bf bb3 = load_b(w3r, k0 + DD, half);
        #pragma unroll
        for (int lt = 0; lt < 2; ++lt) {
            v16bf a = load_a(irow[lt], k0, half);
            c1[lt] = WMMA_BF16(a, bb1, c1[lt]);
            c2[lt] = WMMA_BF16(a, bb2, c2[lt]);
            c3[lt] = WMMA_BF16(a, bb3, c3[lt]);
        }
    }
    const int d = d0 + ln;
    const float bv1 = b1[d], bv2 = b2[d], bv3 = b3[d];
    #pragma unroll
    for (int lt = 0; lt < 2; ++lt) {
        #pragma unroll
        for (int r = 0; r < 8; ++r) {
            size_t idx = ((size_t)b * LL + l0 + 16 * lt + r + 8 * half) * DD + d;
            float xv = x[idx];
            float z  = tanhf(c1[lt][r] + bv1);
            float rg = 1.f / (1.f + __expf(-(c2[lt][r] + bv2)));
            float fg = 1.f / (1.f + __expf(-(c3[lt][r] + bv3)));
            out[idx] = rg * xv + fg * z;
        }
    }
}

extern "C" void kernel_launch(void* const* d_in, const int* in_sizes, int n_in,
                              void* d_out, int out_size, void* d_ws, size_t ws_size,
                              hipStream_t stream)
{
    (void)in_sizes; (void)n_in; (void)out_size; (void)ws_size;
    const float* x   = (const float*)d_in[0];
    const float* wia = (const float*)d_in[1];
    const float* w1  = (const float*)d_in[2];
    const float* w2  = (const float*)d_in[3];
    const float* w3  = (const float*)d_in[4];
    const float* b1  = (const float*)d_in[5];
    const float* b2  = (const float*)d_in[6];
    const float* b3  = (const float*)d_in[7];
    float* out = (float*)d_out;

    char* p = (char*)d_ws;
    auto take = [&](size_t n) { char* r = p; p += (n + 255) & ~(size_t)255; return (void*)r; };
    unsigned short* x_bf   = (unsigned short*)take((size_t)BB * LL * DD * 2);
    unsigned short* xwc_bf = (unsigned short*)take((size_t)BB * LL * DD * 2);
    unsigned short* xT_bf  = (unsigned short*)take((size_t)BB * DD * LL * 2);
    unsigned short* itr_bf = (unsigned short*)take((size_t)BB * LL * DD * 2);
    unsigned short* attn_bf= (unsigned short*)take((size_t)BB * LL * LL * 2);
    unsigned short* w1T    = (unsigned short*)take((size_t)DD * D2 * 2);
    unsigned short* w2T    = (unsigned short*)take((size_t)DD * D2 * 2);
    unsigned short* w3T    = (unsigned short*)take((size_t)DD * D2 * 2);
    float* sa   = (float*)take((size_t)BB * LL * 4);
    float* sb   = (float*)take((size_t)BB * LL * 4);
    float* invl = (float*)take((size_t)BB * LL * 4);

    prep_rows_kernel<<<BB * LL / 8, 256, 0, stream>>>(x, wia, x_bf, xwc_bf, sa, sb);
    transpose_x_kernel<<<dim3(DD / 32, LL / 32, BB), dim3(32, 8), 0, stream>>>(x, xT_bf);
    int wblk = (D2 * DD + 255) / 256;
    prep_w_kernel<<<wblk, 256, 0, stream>>>(w1, w1T);
    prep_w_kernel<<<wblk, 256, 0, stream>>>(w2, w2T);
    prep_w_kernel<<<wblk, 256, 0, stream>>>(w3, w3T);
    attn_softmax_kernel<<<BB * (LL / 16) / 8, 256, 0, stream>>>(x_bf, xwc_bf, sa, sb, attn_bf, invl);
    attn_apply_kernel<<<BB * 7 * 32 / 8, 256, 0, stream>>>(xT_bf, attn_bf, invl, itr_bf);
    mlp_kernel<<<BB * 32 * 28 / 8, 256, 0, stream>>>(
        x, x_bf, itr_bf, w1T, w2T, w3T, b1, b2, b3, out);
}
